// WaveNet_1322849927716
// MI455X (gfx1250) — matrix operand order repaired and checked by reading, not compiled, not run on hardware
//
#include <hip/hip_runtime.h>
#include <hip/hip_bf16.h>

typedef __attribute__((ext_vector_type(16))) __bf16 v16bf;
typedef __attribute__((ext_vector_type(8)))  __bf16 v8bf;
typedef __attribute__((ext_vector_type(8)))  float  v8f;

#define E_LEN   4096
#define CHUNK   2048
#define HALO    2048
#define T_TOTAL 65536
#define NL      18

__constant__ int c_dil[NL]  = {1,2,4,8,16,32,64,128,256,1,2,4,8,16,32,64,128,256};
__constant__ int c_soff[NL] = {0,2,6,14,30,62,126,254,510,1022,1024,1028,1036,1052,1084,1148,1276,1532};

__device__ inline __bf16 f2bf(float f){
  unsigned u = __builtin_bit_cast(unsigned, f);
  unsigned r = u + 0x7FFFu + ((u >> 16) & 1u);      // round-to-nearest-even
  unsigned short h = (unsigned short)(r >> 16);
  return __builtin_bit_cast(__bf16, h);
}
__device__ inline float bf2f(__bf16 b){
  unsigned short h = __builtin_bit_cast(unsigned short, b);
  unsigned u = ((unsigned)h) << 16;
  return __builtin_bit_cast(float, u);
}

// Reorder [O=16][I=16][K=3] weights into per-lane WMMA A-operand layout.
// A_all layout: [layer][op:5][lane:32][16 bf16]; op = {f_tap01, f_tap2, g_tap01, g_tap2, res}
// A k order used here (contiguous; B uses the same): lane<16 row=lane holds K{0..15}; lane>=16 row=lane-16 holds K{16..31}.
__global__ void prep_weights_kernel(const float* __restrict__ conv_w,
                                    const float* __restrict__ gate_w,
                                    const float* __restrict__ res_w,
                                    __bf16* __restrict__ A_all){
  const int li = blockIdx.x;
  const int L  = threadIdx.x;          // 0..31
  const int o  = L & 15;
  const int hi = L >> 4;
  __bf16* base = A_all + ((size_t)(li * 5) * 32 + L) * 16;
  for (int e = 0; e < 16; ++e){
    const int K  = hi * 16 + e;
    const int ch = K & 15;
    const size_t wi = (((size_t)li * 16 + o) * 16 + ch) * 3;
    // WMMA1: K<16 -> tap0 (shift 2d), K>=16 -> tap1 (shift d). WMMA2: K<16 -> tap2, K>=16 -> 0.
    const float f1 = conv_w[wi + (K < 16 ? 0 : 1)];
    const float g1 = gate_w[wi + (K < 16 ? 0 : 1)];
    const float f2 = (K < 16) ? conv_w[wi + 2] : 0.f;
    const float g2 = (K < 16) ? gate_w[wi + 2] : 0.f;
    const float rr = (K < 16 && li < NL - 1) ? res_w[((size_t)li * 16 + o) * 16 + ch] : 0.f;
    base[0 * 512 + e] = f2bf(f1);
    base[1 * 512 + e] = f2bf(f2);
    base[2 * 512 + e] = f2bf(g1);
    base[3 * 512 + e] = f2bf(g2);
    base[4 * 512 + e] = f2bf(rr);
  }
}

__launch_bounds__(256, 1)
__global__ void wavenet_kernel(const float* __restrict__ x,
                               const float* __restrict__ start_w,
                               const float* __restrict__ conv_b,
                               const float* __restrict__ gate_b,
                               const float* __restrict__ res_b,
                               const float* __restrict__ mixer_w,
                               const __bf16* __restrict__ A_all,
                               float* __restrict__ out){
  extern __shared__ char smem[];
  __bf16* hbuf0 = (__bf16*)(smem);                 // [E_LEN*16] bf16 = 128 KB
  __bf16* hbuf1 = (__bf16*)(smem + 131072);        // ping-pong buffer, 128 KB
  __bf16* zbase = (__bf16*)(smem + 262144);        // per-wave z scratch, 8*256 bf16 = 4 KB
  float*  oacc  = (float*) (smem + 266240);        // [E_LEN] f32 skip/mixer accumulator, 16 KB

  const int tid    = threadIdx.x;
  const int lane   = tid & 31;
  const int wave   = tid >> 5;
  const int lane15 = lane & 15;
  const int half   = lane >> 4;
  const int chunk  = blockIdx.x;
  const int b      = blockIdx.y;

  // ---- stage h0 = start_w @ x (bf16, h[t][c] layout), zero accumulator ----
  float sw[16];
  #pragma unroll
  for (int c = 0; c < 16; ++c) sw[c] = start_w[c];
  const long gbase = (long)b * T_TOTAL + (long)chunk * CHUNK - HALO;
  for (int t = tid; t < E_LEN; t += 256){
    const long g = gbase + t;
    const float xv = (g >= 0) ? x[g] : 0.f;   // causal zero pad at sequence start
    #pragma unroll
    for (int c = 0; c < 16; ++c) hbuf0[t * 16 + c] = f2bf(sw[c] * xv);
    oacc[t] = 0.f;
  }
  __syncthreads();

  __bf16* hCur = hbuf0;
  __bf16* hNxt = hbuf1;
  __bf16* zw   = zbase + wave * 256;

  for (int li = 0; li < NL; ++li){
    const int d = c_dil[li];
    const __bf16* abase = A_all + ((size_t)(li * 5) * 32 + lane) * 16;
    const v16bf Af1 = *(const v16bf*)(abase + 0 * 512);
    const v16bf Af2 = *(const v16bf*)(abase + 1 * 512);
    const v16bf Ag1 = *(const v16bf*)(abase + 2 * 512);
    const v16bf Ag2 = *(const v16bf*)(abase + 3 * 512);
    const v16bf Ar  = *(const v16bf*)(abase + 4 * 512);
    const v8f cb = *(const v8f*)(conv_b  + li * 16 + half * 8);
    const v8f gb = *(const v8f*)(gate_b  + li * 16 + half * 8);
    const v8f mw = *(const v8f*)(mixer_w + li * 16 + half * 8);
    v8f rb = {};
    if (li < NL - 1) rb = *(const v8f*)(res_b + li * 16 + half * 8);

    // B operand: lanes 0-15 supply K=0..15 (tap shifted 2d), lanes 16-31 supply K=16..31 (shift d)
    const int shift = (half == 0) ? 2 * d : d;

    for (int tIdx = (c_soff[li] >> 4) + wave; tIdx < E_LEN / 16; tIdx += 8){
      const int col = tIdx * 16 + lane15;                   // this lane's time column
      const v16bf B1 = *(const v16bf*)(hCur + (col - shift) * 16);  // OOB-left -> LDS returns 0 (causal pad)
      const v16bf B2 = *(const v16bf*)(hCur + col * 16);            // tap2 (K upper half of A2 is zero)

      v8f af = cb, ag = gb;
      af = __builtin_amdgcn_wmma_f32_16x16x32_bf16(false, Af1, false, B1, (short)0, af, false, false);
      af = __builtin_amdgcn_wmma_f32_16x16x32_bf16(false, Af2, false, B2, (short)0, af, false, false);
      ag = __builtin_amdgcn_wmma_f32_16x16x32_bf16(false, Ag1, false, B1, (short)0, ag, false, false);
      ag = __builtin_amdgcn_wmma_f32_16x16x32_bf16(false, Ag2, false, B2, (short)0, ag, false, false);

      float part = 0.f;
      v8bf zv;
      #pragma unroll
      for (int r = 0; r < 8; ++r){
        const float fr = af[r], gr = ag[r];
        // fused double-softsign: z = f*g / ((1+|f|)(1+|g|)); denom >= 1 so hw rcp
        // (v_rcp_f32, ~22-bit) is exact enough for a bf16 result — avoids IEEE div expansion.
        const float den = (1.f + fabsf(fr)) * (1.f + fabsf(gr));
        const float z = (fr * gr) * __builtin_amdgcn_rcpf(den);
        part += mw[r] * z;                 // mixer folded in per layer
        zv[r] = f2bf(z);
      }
      atomicAdd(&oacc[col], part);         // two lanes (channel halves) per column

      if (li < NL - 1){
        // Round-trip z through per-wave LDS scratch to re-shape for the residual WMMA B operand.
        *(v8bf*)(zw + lane15 * 16 + half * 8) = zv;
        asm volatile("s_wait_dscnt 0x0" ::: "memory");       // intra-wave LDS RAW fence
        const v16bf Bz  = *(const v16bf*)(zw + lane15 * 16);
        const v8bf hold = *(const v8bf*)(hCur + col * 16 + half * 8);
        v8f ah = rb;
        #pragma unroll
        for (int r = 0; r < 8; ++r) ah[r] += bf2f(hold[r]);
        ah = __builtin_amdgcn_wmma_f32_16x16x32_bf16(false, Ar, false, Bz, (short)0, ah, false, false);
        v8bf hn;
        #pragma unroll
        for (int r = 0; r < 8; ++r) hn[r] = f2bf(ah[r]);
        *(v8bf*)(hNxt + col * 16 + half * 8) = hn;
      }
    }
    __syncthreads();
    __bf16* tmp = hCur; hCur = hNxt; hNxt = tmp;
  }

  // ---- write the valid (non-halo) half of the accumulator ----
  for (int t = HALO + tid; t < E_LEN; t += 256){
    out[(long)b * T_TOTAL + (long)chunk * CHUNK + (t - HALO)] = oacc[t];
  }
}

extern "C" void kernel_launch(void* const* d_in, const int* in_sizes, int n_in,
                              void* d_out, int out_size, void* d_ws, size_t ws_size,
                              hipStream_t stream) {
  const float* x       = (const float*)d_in[0];
  const float* start_w = (const float*)d_in[1];
  const float* conv_w  = (const float*)d_in[2];
  const float* conv_b  = (const float*)d_in[3];
  const float* gate_w  = (const float*)d_in[4];
  const float* gate_b  = (const float*)d_in[5];
  const float* res_w   = (const float*)d_in[6];
  const float* res_b   = (const float*)d_in[7];
  const float* mixer_w = (const float*)d_in[8];
  float* out = (float*)d_out;
  __bf16* A_all = (__bf16*)d_ws;   // 18*5*32*16 bf16 = 92160 bytes of scratch

  prep_weights_kernel<<<dim3(NL), dim3(32), 0, stream>>>(conv_w, gate_w, res_w, A_all);

  // 32 time-chunks x 4 batch, 256 threads (8 wave32), 276 KB dynamic LDS (of 320 KB WGP budget)
  wavenet_kernel<<<dim3(T_TOTAL / CHUNK, 4), dim3(256), 282624, stream>>>(
      x, start_w, conv_b, gate_b, res_b, mixer_w, A_all, out);
}
